// SpatialSTEncoder_87333864997251
// MI455X (gfx1250) — matrix operand order repaired and checked by reading, not compiled, not run on hardware
//
#include <hip/hip_runtime.h>
#include <math.h>
#include <stdint.h>

// ---------------- problem constants ----------------
#define N_    128
#define GK    2048          // gene tokens
#define DD    256           // model dim
#define L_    2
#define H_    4
#define DH_   64
#define MFEAT 266           // performer feature dim
#define MF    288           // padded to mult of 32
#define SEQ   2049          // 1 spatial + 2048 genes
#define SP    2080          // padded to mult of 32
#define RRI   (N_*SP)       // 266240 total rows

#define CS2   0.125f                    // DH^-0.5
#define CQ    0.35355339059327373f      // DH^-0.25
#define RATIO 0.06131393394849658f      // 1/sqrt(266)
#define KEPS  1.0e-4f

typedef __attribute__((ext_vector_type(16))) _Float16 v16h;
typedef __attribute__((ext_vector_type(8)))  _Float16 v8h;
typedef __attribute__((ext_vector_type(8)))  float    v8f;
typedef _Float16 f16;

// ---------------- block reduction helpers (blockDim == 256) ----------------
__device__ __forceinline__ float block_sum(float v, float* sm) {
    int t = threadIdx.x;
    sm[t] = v; __syncthreads();
    for (int off = 128; off > 0; off >>= 1) {
        if (t < off) sm[t] += sm[t + off];
        __syncthreads();
    }
    float r = sm[0]; __syncthreads();
    return r;
}
__device__ __forceinline__ float block_max(float v, float* sm) {
    int t = threadIdx.x;
    sm[t] = v; __syncthreads();
    for (int off = 128; off > 0; off >>= 1) {
        if (t < off) sm[t] = fmaxf(sm[t], sm[t + off]);
        __syncthreads();
    }
    float r = sm[0]; __syncthreads();
    return r;
}
__device__ __forceinline__ float gelu_f(float v) {
    return 0.5f * v * (1.0f + erff(v * 0.7071067811865476f));
}
__device__ __forceinline__ void wait_async0() {
#if __has_builtin(__builtin_amdgcn_s_wait_asynccnt)
    __builtin_amdgcn_s_wait_asynccnt(0);
#else
    asm volatile("s_wait_asynccnt 0x0" ::: "memory");
#endif
}
__device__ __forceinline__ void async_b128(unsigned lds_off, const void* gptr) {
    asm volatile("global_load_async_to_lds_b128 %0, %1, off"
                 :: "v"(lds_off), "v"((unsigned long long)(uintptr_t)gptr)
                 : "memory");
}

// ---------------- generic batched WMMA GEMM ----------------
// C[b] = A[b](MxK,f16) * B[b](KxN,f16)  (+bias) (gelu?) (*rowscale?) (+=residual?)
// optional f32 out (C) and/or f16 out (D16).  Batch offsets: b1=b/bh, b2=b%bh.
// Tile BMxBNx32; waves arranged (BM/(16*RF)) x (BN/(16*CF)); NT threads.
// Double-buffered LDS: slab k+32 staged (A via GLOBAL_LOAD_ASYNC_TO_LDS_B128,
// B via load + transposed ds_store) while slab k feeds RF*CF WMMAs per wave.
template<int BM, int BN, int RF, int CF, int NT>
__global__ __launch_bounds__(NT) void wmma_gemm_kernel(
    const f16* __restrict__ A, long sA1, long sA2, int lda,
    const f16* __restrict__ B, long sB1, long sB2, int ldb,
    float* C, long sC1, long sC2, int ldc,
    f16* D16, long sD1, long sD2, int ldd,
    const float* __restrict__ bias,
    const float* __restrict__ rowscale, long sR1, long sR2,
    int M, int Ncols, int Kdim, int bh, int flags)   // flags: 1=gelu, 2=residual add
{
    constexpr int WROWS = BM / (16 * RF);
    constexpr int WCOLS = BN / (16 * CF);
    static_assert(WROWS * WCOLS == NT / 32, "wave layout mismatch");
    constexpr int ACH = (BM * 4) / NT;   // 8-half A chunks per thread
    constexpr int BCH = (BN * 4) / NT;   // 8-half B chunks per thread
    constexpr int CPR = BN / 8;          // B chunks per k-row

    __shared__ f16 As[2][BM][40];        // BM rows x 32 K (+8 pad), double-buffered
    __shared__ f16 Bs[2][BN][40];        // transposed: [col][k]

    const int lane = threadIdx.x & 31;
    const int wave = threadIdx.x >> 5;
    const int wr   = wave / WCOLS;
    const int wc   = wave % WCOLS;

    long b  = blockIdx.z;
    long b1 = b / bh, b2 = b - b1 * bh;
    const f16* Ab = A + b1 * sA1 + b2 * sA2;
    const f16* Bb = B + b1 * sB1 + b2 * sB2;
    long offC = b1 * sC1 + b2 * sC2;
    long offD = b1 * sD1 + b2 * sD2;
    long offR = b1 * sR1 + b2 * sR2;

    const int rowBase = blockIdx.y * BM;
    const int colBase = blockIdx.x * BN;
    const bool fullM  = (rowBase + BM <= M);
    const bool fullN  = (colBase + BN <= Ncols);

    // stage one 32-K slab into buffer `buf`
    auto stage = [&](int buf, int k0) {
        // ---- A tile (BMx32): per-lane contiguous 16B -> async to LDS ----
        if (fullM) {
            #pragma unroll
            for (int i = 0; i < ACH; ++i) {
                int c  = threadIdx.x * ACH + i;
                int ar = c >> 2;
                int ac = (c & 3) << 3;
                async_b128((unsigned)(uintptr_t)&As[buf][ar][ac],
                           Ab + (long)(rowBase + ar) * lda + (k0 + ac));
            }
        } else {
            #pragma unroll
            for (int i = 0; i < ACH; ++i) {
                int c  = threadIdx.x * ACH + i;
                int ar = c >> 2;
                int ac = (c & 3) << 3;
                if (rowBase + ar < M) {
                    async_b128((unsigned)(uintptr_t)&As[buf][ar][ac],
                               Ab + (long)(rowBase + ar) * lda + (k0 + ac));
                } else {
                    v8h z = {};
                    *(v8h*)&As[buf][ar][ac] = z;
                }
            }
        }
        // ---- B tile (32xBN), stored transposed [col][k] ----
        #pragma unroll
        for (int i = 0; i < BCH; ++i) {
            int c  = threadIdx.x * BCH + i;
            int br = c / CPR;               // k row 0..31
            int bc = (c % CPR) * 8;         // col chunk
            v8h bv = {};
            if (fullN || colBase + bc < Ncols)
                bv = *(const v8h*)(Bb + (long)(k0 + br) * ldb + colBase + bc);
            #pragma unroll
            for (int j = 0; j < 8; ++j) Bs[buf][bc + j][br] = bv[j];
        }
    };

    v8f acc[RF][CF] = {};

    stage(0, 0);
    wait_async0();
    __syncthreads();

    int cur = 0;
    for (int k0 = 0; k0 < Kdim; k0 += 32) {
        int nxt = cur ^ 1;
        if (k0 + 32 < Kdim) stage(nxt, k0 + 32);           // overlap with compute
        if (k0 + 64 < Kdim)                                 // deep prefetch of B
            __builtin_prefetch((const void*)(Bb + (long)(k0 + 64 + (threadIdx.x >> 3)) * ldb + colBase), 0, 1);

        // ---- A fragments (16x32 each): lanes 0-15 K0..7/16..23, lanes 16-31 K8..15/24..31
        v16h afrag[RF];
        #pragma unroll
        for (int rf = 0; rf < RF; ++rf) {
            int am  = (wr * RF + rf) * 16 + (lane & 15);
            int akb = (lane >> 4) << 3;
            v8h alo = *(const v8h*)&As[cur][am][akb];
            v8h ahi = *(const v8h*)&As[cur][am][akb + 16];
            #pragma unroll
            for (int q = 0; q < 8; ++q) { afrag[rf][q] = alo[q]; afrag[rf][q + 8] = ahi[q]; }
        }
        // ---- B fragments (32x16): lanes 0-15 K0..7/16..23, lanes 16-31 K8..15/24..31
        #pragma unroll
        for (int cf = 0; cf < CF; ++cf) {
            int bn  = (wc * CF + cf) * 16 + (lane & 15);
            int bkb = (lane >> 4) << 3;
            v8h blo = *(const v8h*)&Bs[cur][bn][bkb];
            v8h bhi = *(const v8h*)&Bs[cur][bn][bkb + 16];
            v16h bfrag;
            #pragma unroll
            for (int q = 0; q < 8; ++q) { bfrag[q] = blo[q]; bfrag[q + 8] = bhi[q]; }
            #pragma unroll
            for (int rf = 0; rf < RF; ++rf) {
                acc[rf][cf] = __builtin_amdgcn_wmma_f32_16x16x32_f16(
                    false, afrag[rf], false, bfrag, (short)0, acc[rf][cf], false, false);
            }
        }
        wait_async0();       // next slab's async copies done before barrier
        __syncthreads();
        cur = nxt;
    }

    // epilogue:  C/D layout: lanes 0-15 -> rows +0..7, lanes 16-31 -> rows +8..15
    #pragma unroll
    for (int cf = 0; cf < CF; ++cf) {
        int gcol = colBase + (wc * CF + cf) * 16 + (lane & 15);
        if (gcol >= Ncols) continue;
        #pragma unroll
        for (int rf = 0; rf < RF; ++rf) {
            #pragma unroll
            for (int e = 0; e < 8; ++e) {
                int grow = rowBase + (wr * RF + rf) * 16 + ((lane >> 4) << 3) + e;
                if (grow >= M) continue;
                float r = acc[rf][cf][e];
                if (bias)       r += bias[gcol];
                if (flags & 1)  r = gelu_f(r);
                if (rowscale)   r *= rowscale[offR + grow];
                if (C) {
                    long ci = offC + (long)grow * ldc + gcol;
                    if (flags & 2) C[ci] += r; else C[ci] = r;
                }
                if (D16) D16[offD + (long)grow * ldd + gcol] = (f16)r;
            }
        }
    }
}

// ---------------- elementwise / reduction kernels ----------------
__global__ void cvt_f16_kernel(const float* __restrict__ s, f16* __restrict__ d, long n) {
    long i = (long)blockIdx.x * blockDim.x + threadIdx.x;
    if (i < n) d[i] = (f16)s[i];
}

// projT16[l][d][m] = proj[l][m][d] * DH^-0.25  (zero-padded m>=266)
__global__ void proj_cvt_kernel(const float* __restrict__ proj, f16* __restrict__ out) {
    long i = (long)blockIdx.x * blockDim.x + threadIdx.x;
    long tot = (long)L_ * DH_ * MF;
    if (i >= tot) return;
    int m = (int)(i % MF);
    long r = i / MF;
    int d2 = (int)(r % DH_);
    int l  = (int)(r / DH_);
    float v = 0.f;
    if (m < MFEAT) v = proj[((long)l * MFEAT + m) * DH_ + d2] * CQ;
    out[i] = (f16)v;
}

// x[n,s,:] token embedding (spatial token at s=0, genes at s=1..2048, pad rows zero)
__global__ void build_tokens_kernel(
    const float* __restrict__ expr, const float* __restrict__ coords,
    const float* __restrict__ gene_emb, const float* __restrict__ gene_pos,
    const float* __restrict__ val_w, const float* __restrict__ val_b,
    const float* __restrict__ vg, const float* __restrict__ vb,
    const float* __restrict__ sp_w, const float* __restrict__ sp_b,
    const float* __restrict__ sg, const float* __restrict__ sb,
    float* __restrict__ x)
{
    __shared__ float sm[256];
    int n = blockIdx.y, s = blockIdx.x, t = threadIdx.x;
    long xoff = ((long)n * SP + s) * DD + t;
    if (s >= SEQ) { x[xoff] = 0.f; return; }
    float pre;
    const float *g, *b;
    if (s == 0) {
        pre = coords[n*2] * sp_w[t] + coords[n*2+1] * sp_w[DD + t] + sp_b[t];
        g = sg; b = sb;
    } else {
        int k = s - 1;
        pre = expr[(long)n * GK + k] * val_w[t] + val_b[t];
        g = vg; b = vb;
    }
    float mu  = block_sum(pre, sm) * (1.f / DD);
    float d0  = pre - mu;
    float var = block_sum(d0 * d0, sm) * (1.f / DD);
    float y   = gelu_f(d0 * rsqrtf(var + 1e-5f) * g[t] + b[t]);
    if (s > 0) y += gene_emb[(long)(s-1) * DD + t] + gene_pos[(long)(s-1) * DD + t];
    x[xoff] = y;
}

// out16 = f16( LN(x_row) * g + b ),  zero for padded rows
__global__ void ln_affine_kernel(const float* __restrict__ x, f16* __restrict__ out,
                                 const float* __restrict__ g, const float* __restrict__ b)
{
    __shared__ float sm[256];
    long row = blockIdx.x;
    int s = (int)(row % SP);
    int t = threadIdx.x;
    long off = row * DD + t;
    if (s >= SEQ) { out[off] = (f16)0.f; return; }
    float v   = x[off];
    float mu  = block_sum(v, sm) * (1.f / DD);
    float d0  = v - mu;
    float var = block_sum(d0 * d0, sm) * (1.f / DD);
    out[off] = (f16)(d0 * rsqrtf(var + 1e-5f) * g[t] + b[t]);
}

// per (n,h,s): diag = 0.5*c^2*sum(q^2) ; rowmax over m<266 of dd
__global__ void rowstats_kernel(const float* __restrict__ dd, const f16* __restrict__ src,
                                float* __restrict__ diag, float* __restrict__ rmax)
{
    __shared__ float sm[256];
    long ridx = blockIdx.x;          // nh*SP + s
    int  s    = (int)(ridx % SP);
    long nh   = ridx / SP;
    int  h    = (int)(nh % H_);
    long n    = nh / H_;
    int  t    = threadIdx.x;
    float sq = 0.f;
    if (t < DH_) {
        float qv = (float)src[(n * SP + s) * (long)DD + h * DH_ + t];
        sq = qv * qv;
    }
    float sum = block_sum(sq, sm);
    float mx = -1e30f;
    const float* row = dd + ridx * MF;
    for (int m = t; m < MFEAT; m += 256) mx = fmaxf(mx, row[m]);
    mx = block_max(mx, sm);
    if (t == 0) {
        diag[ridx] = 0.5f * CS2 * sum;
        rmax[ridx] = (s < SEQ) ? mx : -1e30f;
    }
}

__global__ void gmax_kernel(const float* __restrict__ rmax, float* __restrict__ gmx) {
    __shared__ float sm[256];
    long nh = blockIdx.x; int t = threadIdx.x;
    float mx = -1e30f;
    for (int s = t; s < SEQ; s += 256) mx = fmaxf(mx, rmax[nh * SP + s]);
    mx = block_max(mx, sm);
    if (t == 0) gmx[nh] = mx;
}

// qp (row-major) or kp (transposed [m][s]) = ratio*(exp(dd - diag - mx)+eps), zero padded
__global__ void featmap_kernel(const float* __restrict__ dd, const float* __restrict__ diag,
                               const float* __restrict__ rmax, const float* __restrict__ gmx,
                               f16* __restrict__ out, int isQuery)
{
    long ridx = blockIdx.x;
    int  s    = (int)(ridx % SP);
    long nh   = ridx / SP;
    int  t    = threadIdx.x;
    float mx = isQuery ? rmax[ridx] : gmx[nh];
    float dg = diag[ridx];
    const float* row = dd + ridx * MF;
    for (int m = t; m < MF; m += 256) {
        float v = 0.f;
        if (s < SEQ && m < MFEAT) v = RATIO * (expf(row[m] - dg - mx) + KEPS);
        if (isQuery) out[ridx * MF + m] = (f16)v;
        else         out[(nh * MF + m) * SP + s] = (f16)v;
    }
}

__global__ void ksum_kernel(const f16* __restrict__ kpT, float* __restrict__ ks) {
    __shared__ float sm[256];
    long idx = blockIdx.x;          // nh*MF + m
    int t = threadIdx.x;
    float sum = 0.f;
    const f16* row = kpT + idx * SP;
    for (int s = t; s < SP; s += 256) sum += (float)row[s];
    sum = block_sum(sum, sm);
    if (t == 0) ks[idx] = sum;
}

__global__ void dinv_kernel(const f16* __restrict__ qp, const float* __restrict__ ks,
                            float* __restrict__ dv)
{
    __shared__ float sm[256];
    long ridx = blockIdx.x;
    int  s    = (int)(ridx % SP);
    long nh   = ridx / SP;
    int  t    = threadIdx.x;
    float sum = 0.f;
    for (int m = t; m < MF; m += 256) sum += (float)qp[ridx * MF + m] * ks[nh * MF + m];
    sum = block_sum(sum, sm);
    if (t == 0) dv[ridx] = (s < SEQ) ? 1.f / sum : 0.f;
}

// final pooling: per cell softmax(q.k/16) over 2048 genes, weighted sum of v
__global__ void pool_attend_kernel(const f16* __restrict__ q, const f16* __restrict__ kk,
                                   const f16* __restrict__ vv,
                                   float* __restrict__ pooled, f16* __restrict__ pooled16)
{
    __shared__ float logits[GK];
    __shared__ float sm[256];
    int n = blockIdx.x, t = threadIdx.x;
    const f16* qr = q + (long)n * DD;
    for (int k = t; k < GK; k += 256) {
        const f16* kr = kk + ((long)n * GK + k) * DD;
        float dot = 0.f;
        for (int d2 = 0; d2 < DD; ++d2) dot += (float)qr[d2] * (float)kr[d2];
        logits[k] = dot * (1.f / 16.f);          // / sqrt(256)
    }
    __syncthreads();
    float mx = -1e30f;
    for (int k = t; k < GK; k += 256) mx = fmaxf(mx, logits[k]);
    mx = block_max(mx, sm);
    float sum = 0.f;
    for (int k = t; k < GK; k += 256) { float e = expf(logits[k] - mx); logits[k] = e; sum += e; }
    sum = block_sum(sum, sm);
    float inv = 1.f / sum;
    __syncthreads();
    float acc = 0.f;
    for (int k = 0; k < GK; ++k) acc += logits[k] * inv * (float)vv[((long)n * GK + k) * DD + t];
    pooled[(long)n * DD + t]   = acc;
    pooled16[(long)n * DD + t] = (f16)acc;
}

// ---------------- host orchestration ----------------
extern "C" void kernel_launch(void* const* d_in, const int* in_sizes, int n_in,
                              void* d_out, int out_size, void* d_ws, size_t ws_size,
                              hipStream_t stream)
{
    (void)in_sizes; (void)n_in; (void)out_size; (void)ws_size;

    const float* expr    = (const float*)d_in[0];
    const float* coords  = (const float*)d_in[1];
    const float* gene_emb= (const float*)d_in[2];
    const float* gene_pos= (const float*)d_in[3];
    const float* val_w   = (const float*)d_in[4];
    const float* val_b   = (const float*)d_in[5];
    const float* val_ln_g= (const float*)d_in[6];
    const float* val_ln_b= (const float*)d_in[7];
    const float* sp_w    = (const float*)d_in[8];
    const float* sp_b    = (const float*)d_in[9];
    const float* sp_ln_g = (const float*)d_in[10];
    const float* sp_ln_b = (const float*)d_in[11];
    const float* ln1_g   = (const float*)d_in[12];
    const float* ln1_b   = (const float*)d_in[13];
    const float* wq      = (const float*)d_in[14];
    const float* bq      = (const float*)d_in[15];
    const float* wk      = (const float*)d_in[16];
    const float* bk      = (const float*)d_in[17];
    const float* wv      = (const float*)d_in[18];
    const float* bv      = (const float*)d_in[19];
    const float* wo      = (const float*)d_in[20];
    const float* bo      = (const float*)d_in[21];
    const float* ln2_g   = (const float*)d_in[22];
    const float* ln2_b   = (const float*)d_in[23];
    const float* w1      = (const float*)d_in[24];
    const float* b1      = (const float*)d_in[25];
    const float* w2      = (const float*)d_in[26];
    const float* b2      = (const float*)d_in[27];
    const float* pq_w    = (const float*)d_in[28];
    const float* pq_b    = (const float*)d_in[29];
    const float* pk_w    = (const float*)d_in[30];
    const float* pk_b    = (const float*)d_in[31];
    const float* pv_w    = (const float*)d_in[32];
    const float* pv_b    = (const float*)d_in[33];
    const float* po_w    = (const float*)d_in[34];
    const float* po_b    = (const float*)d_in[35];
    const float* proj    = (const float*)d_in[36];

    // ---- workspace bump allocator ----
    char* ws = (char*)d_ws;
    size_t off = 0;
    auto alloc = [&](size_t bytes) -> char* {
        char* p = ws + off;
        off = (off + bytes + 255) & ~(size_t)255;
        return p;
    };
    const long RD  = (long)RRI * DD;
    const long NHS = (long)N_ * H_ * SP;

    float* x      = (float*)alloc(RD * 4);
    f16*   h16    = (f16*)  alloc(RD * 2);
    f16*   q16    = (f16*)  alloc(RD * 2);
    f16*   k16    = (f16*)  alloc(RD * 2);
    f16*   v16    = (f16*)  alloc(RD * 2);
    f16*   att16  = (f16*)  alloc(RD * 2);
    f16*   x16    = (f16*)  alloc(RD * 2);
    f16*   gf16   = (f16*)  alloc((long)RRI * 1024 * 2);
    float* ddb    = (float*)alloc(NHS * MF * 4);
    float* diagb  = (float*)alloc(NHS * 4);
    float* rmaxb  = (float*)alloc(NHS * 4);
    float* gmxb   = (float*)alloc((long)N_ * H_ * 4);
    f16*   qp16   = (f16*)  alloc(NHS * MF * 2);
    f16*   kpT16  = (f16*)  alloc(NHS * MF * 2);
    float* ksb    = (float*)alloc((long)N_ * H_ * MF * 4);
    float* dvb    = (float*)alloc(NHS * 4);
    f16*   ctx16  = (f16*)  alloc((long)N_ * H_ * MF * DH_ * 2);
    f16*   kk16   = (f16*)  alloc((long)N_ * GK * DD * 2);
    f16*   vv16   = (f16*)  alloc((long)N_ * GK * DD * 2);
    f16*   qpool16= (f16*)  alloc((long)N_ * DD * 2);
    float* pooled = (float*)alloc((long)N_ * DD * 4);
    f16*   pooled16=(f16*)  alloc((long)N_ * DD * 2);
    f16*   wq16   = (f16*)  alloc((long)L_ * DD * DD * 2);
    f16*   wk16   = (f16*)  alloc((long)L_ * DD * DD * 2);
    f16*   wv16   = (f16*)  alloc((long)L_ * DD * DD * 2);
    f16*   wo16   = (f16*)  alloc((long)L_ * DD * DD * 2);
    f16*   w116   = (f16*)  alloc((long)L_ * DD * 1024 * 2);
    f16*   w216   = (f16*)  alloc((long)L_ * 1024 * DD * 2);
    f16*   projT16= (f16*)  alloc((long)L_ * DH_ * MF * 2);
    f16*   pqw16  = (f16*)  alloc((long)DD * DD * 2);
    f16*   pkw16  = (f16*)  alloc((long)DD * DD * 2);
    f16*   pvw16  = (f16*)  alloc((long)DD * DD * 2);
    f16*   pow16  = (f16*)  alloc((long)DD * DD * 2);

    auto cvt = [&](const float* s, f16* d, long n) {
        unsigned blocks = (unsigned)((n + 255) / 256);
        cvt_f16_kernel<<<blocks, 256, 0, stream>>>(s, d, n);
    };
    auto gemm = [&](const f16* A, long sA1, long sA2, int lda,
                    const f16* B, long sB1, long sB2, int ldb,
                    float* C, long sC1, long sC2, int ldc,
                    f16* D16, long sD1, long sD2, int ldd,
                    const float* bias, const float* rs, long sR1, long sR2,
                    int M, int Ncols, int Kd, int batch, int bh, int flags) {
        if (Ncols >= 128 && M >= 128) {
            dim3 grid((Ncols + 127) / 128, (M + 127) / 128, batch);
            wmma_gemm_kernel<128,128,2,4,256><<<grid, dim3(256), 0, stream>>>(
                A, sA1, sA2, lda, B, sB1, sB2, ldb, C, sC1, sC2, ldc,
                D16, sD1, sD2, ldd, bias, rs, sR1, sR2, M, Ncols, Kd, bh, flags);
        } else {
            dim3 grid((Ncols + 63) / 64, (M + 63) / 64, batch);
            wmma_gemm_kernel<64,64,1,4,128><<<grid, dim3(128), 0, stream>>>(
                A, sA1, sA2, lda, B, sB1, sB2, ldb, C, sC1, sC2, ldc,
                D16, sD1, sD2, ldd, bias, rs, sR1, sR2, M, Ncols, Kd, bh, flags);
        }
    };

    // ---- weight conversion f32 -> f16 ----
    cvt(wq, wq16, (long)L_ * DD * DD);
    cvt(wk, wk16, (long)L_ * DD * DD);
    cvt(wv, wv16, (long)L_ * DD * DD);
    cvt(wo, wo16, (long)L_ * DD * DD);
    cvt(w1, w116, (long)L_ * DD * 1024);
    cvt(w2, w216, (long)L_ * 1024 * DD);
    cvt(pq_w, pqw16, (long)DD * DD);
    cvt(pk_w, pkw16, (long)DD * DD);
    cvt(pv_w, pvw16, (long)DD * DD);
    cvt(po_w, pow16, (long)DD * DD);
    {
        long tot = (long)L_ * DH_ * MF;
        proj_cvt_kernel<<<(unsigned)((tot + 255) / 256), 256, 0, stream>>>(proj, projT16);
    }

    // ---- token embedding ----
    build_tokens_kernel<<<dim3(SP, N_), 256, 0, stream>>>(
        expr, coords, gene_emb, gene_pos, val_w, val_b, val_ln_g, val_ln_b,
        sp_w, sp_b, sp_ln_g, sp_ln_b, x);

    // ---- transformer layers ----
    for (int l = 0; l < L_; ++l) {
        const f16* wql = wq16 + (long)l * DD * DD;
        const f16* wkl = wk16 + (long)l * DD * DD;
        const f16* wvl = wv16 + (long)l * DD * DD;
        const f16* wol = wo16 + (long)l * DD * DD;
        const f16* w1l = w116 + (long)l * DD * 1024;
        const f16* w2l = w216 + (long)l * 1024 * DD;
        const f16* pjl = projT16 + (long)l * DH_ * MF;

        // ln1 -> h16
        ln_affine_kernel<<<RRI, 256, 0, stream>>>(x, h16, ln1_g + (long)l * DD, ln1_b + (long)l * DD);

        // q,k,v = h @ W + b   (f16 outputs)
        gemm(h16,0,0,DD,  wql,0,0,DD,  nullptr,0,0,0,  q16,0,0,DD,  bq + (long)l*DD, nullptr,0,0, RRI,DD,DD, 1,1, 0);
        gemm(h16,0,0,DD,  wkl,0,0,DD,  nullptr,0,0,0,  k16,0,0,DD,  bk + (long)l*DD, nullptr,0,0, RRI,DD,DD, 1,1, 0);
        gemm(h16,0,0,DD,  wvl,0,0,DD,  nullptr,0,0,0,  v16,0,0,DD,  bv + (long)l*DD, nullptr,0,0, RRI,DD,DD, 1,1, 0);

        // --- query features ---
        gemm(q16,(long)SP*DD,DH_,DD,  pjl,0,0,MF,
             ddb,(long)H_*SP*MF,(long)SP*MF,MF,  nullptr,0,0,0,
             nullptr, nullptr,0,0, SP,MF,DH_, N_*H_, H_, 0);
        rowstats_kernel<<<(unsigned)NHS, 256, 0, stream>>>(ddb, q16, diagb, rmaxb);
        featmap_kernel <<<(unsigned)NHS, 256, 0, stream>>>(ddb, diagb, rmaxb, gmxb, qp16, 1);

        // --- key features (transposed store) ---
        gemm(k16,(long)SP*DD,DH_,DD,  pjl,0,0,MF,
             ddb,(long)H_*SP*MF,(long)SP*MF,MF,  nullptr,0,0,0,
             nullptr, nullptr,0,0, SP,MF,DH_, N_*H_, H_, 0);
        rowstats_kernel<<<(unsigned)NHS, 256, 0, stream>>>(ddb, k16, diagb, rmaxb);
        gmax_kernel    <<<N_*H_, 256, 0, stream>>>(rmaxb, gmxb);
        featmap_kernel <<<(unsigned)NHS, 256, 0, stream>>>(ddb, diagb, rmaxb, gmxb, kpT16, 0);

        // ksum[nh][m] = sum_s kp
        ksum_kernel<<<N_*H_*MF, 256, 0, stream>>>(kpT16, ksb);

        // ctx[nh] = kp^T (MF x SP) @ v_head (SP x 64)
        gemm(kpT16,(long)H_*MF*SP,(long)MF*SP,SP,  v16,(long)SP*DD,DH_,DD,
             nullptr,0,0,0,  ctx16,(long)H_*MF*DH_,(long)MF*DH_,DH_,
             nullptr, nullptr,0,0, MF,DH_,SP, N_*H_, H_, 0);

        // d_inv
        dinv_kernel<<<(unsigned)NHS, 256, 0, stream>>>(qp16, ksb, dvb);

        // att = (qp @ ctx) * d_inv  -> att16[n,s,h*64+..]
        gemm(qp16,(long)H_*SP*MF,(long)SP*MF,MF,  ctx16,(long)H_*MF*DH_,(long)MF*DH_,DH_,
             nullptr,0,0,0,  att16,(long)SP*DD,DH_,DD,
             nullptr, dvb,(long)H_*SP,(long)SP, SP,DH_,MF, N_*H_, H_, 0);

        // x += att @ wo + bo
        gemm(att16,0,0,DD,  wol,0,0,DD,  x,0,0,DD,  nullptr,0,0,0,
             bo + (long)l*DD, nullptr,0,0, RRI,DD,DD, 1,1, 2);

        // FFN: x += gelu(ln2(x) @ w1 + b1) @ w2 + b2
        ln_affine_kernel<<<RRI, 256, 0, stream>>>(x, h16, ln2_g + (long)l * DD, ln2_b + (long)l * DD);
        gemm(h16,0,0,DD,   w1l,0,0,1024,  nullptr,0,0,0,  gf16,0,0,1024,
             b1 + (long)l*1024, nullptr,0,0, RRI,1024,DD, 1,1, 1);
        gemm(gf16,0,0,1024, w2l,0,0,DD,   x,0,0,DD,  nullptr,0,0,0,
             b2 + (long)l*DD,  nullptr,0,0, RRI,DD,1024, 1,1, 2);
    }

    // ---- attention pooling head ----
    cvt(x, x16, RD);
    // q = x[:,0,:] @ pq_w + pq_b   (row stride SP*DD picks s=0 rows)
    gemm(x16,0,0,SP*DD,  pqw16,0,0,DD,  nullptr,0,0,0,  qpool16,0,0,DD,
         pq_b, nullptr,0,0, N_,DD,DD, 1,1, 0);
    // kk / vv over gene tokens (s=1..2048), batched per cell
    gemm(x16+DD,(long)SP*DD,0,DD,  pkw16,0,0,DD,  nullptr,0,0,0,
         kk16,(long)GK*DD,0,DD,  pk_b, nullptr,0,0, GK,DD,DD, N_,1, 0);
    gemm(x16+DD,(long)SP*DD,0,DD,  pvw16,0,0,DD,  nullptr,0,0,0,
         vv16,(long)GK*DD,0,DD,  pv_b, nullptr,0,0, GK,DD,DD, N_,1, 0);

    pool_attend_kernel<<<N_, 256, 0, stream>>>(qpool16, kk16, vv16, pooled, pooled16);

    // out = pooled @ po_w + po_b
    gemm(pooled16,0,0,DD,  pow16,0,0,DD,  (float*)d_out,0,0,DD,  nullptr,0,0,0,
         po_b, nullptr,0,0, N_,DD,DD, 1,1, 0);
}
